// SpatialAttentionLayer_50577534877870
// MI455X (gfx1250) — matrix operand order repaired and checked by reading, not compiled, not run on hardware
//
#include <hip/hip_runtime.h>

// Problem constants (match reference)
#define B_   8
#define N_   1024
#define C_   512
#define H_   8
#define D_   64
#define BH_  (B_ * H_)        // 64
#define NEGV (-10000.0f)
#define SPAD 1040             // LDS fp32 row stride (1024 + 16): conflict-free softmax
#define APAD 520              // LDS f16 A-tile row stride (512 + 8): conflict-free ds frags

typedef __attribute__((ext_vector_type(16))) _Float16 v16h;
typedef __attribute__((ext_vector_type(8)))  _Float16 v8h;
typedef __attribute__((ext_vector_type(8)))  float    v8f;
typedef __attribute__((ext_vector_type(4)))  unsigned v4u;
typedef __attribute__((ext_vector_type(8)))  int      v8i;
typedef __attribute__((ext_vector_type(4)))  int      v4i;

static __device__ __forceinline__ v8f zero_v8f() {
  v8f z = {0.f, 0.f, 0.f, 0.f, 0.f, 0.f, 0.f, 0.f};
  return z;
}

static __device__ __forceinline__ v8f wmma_f16(v16h a, v16h b, v8f c) {
  return __builtin_amdgcn_wmma_f32_16x16x32_f16(false, a, false, b, (short)0, c,
                                                false, false);
}

// A fragment (16x32 f16): per lane row=(lane&15); elem j -> K=k0+j,
// k0 = ktile*32 + (lane>>4)*16.  Ptr pre-offset.
static __device__ __forceinline__ v16h load_a_f16(const _Float16* p) {
  v8h lo = *(const v8h*)p;
  v8h hi = *(const v8h*)(p + 8);
  v16h a;
#pragma unroll
  for (int j = 0; j < 8; ++j) { a[j] = lo[j]; a[j + 8] = hi[j]; }
  return a;
}
// A fragment from fp32 source (cvt inline) -- used for LDS-resident P.
static __device__ __forceinline__ v16h load_a_f32(const float* p) {
  v16h a;
#pragma unroll
  for (int j = 0; j < 8; ++j) {
    a[j]     = (_Float16)p[j];
    a[j + 8] = (_Float16)p[j + 8];
  }
  return a;
}
// B fragment (32x16 f16): per lane col=(lane&15); elem j -> K=k0+(lane>>4)*16+j.
static __device__ __forceinline__ v16h load_b_f16(const _Float16* p) {
  return *(const v16h*)p;
}

// ---------------------------------------------------------------------------
// TDM: 2D tile (rows x d0 f16 elements, row-major, global stride = d0) into
// LDS at lds_off, with LDS padding of 4 DWORDs (8 halves) after every 256
// DWORDs (= one 512-element row) -> LDS row stride APAD = 520 halves.
// D# bitfields per CDNA5 ISA.
// Toolchain uses the 6-arg builtin: (v4u, v8i, v4i, v4i, v8i, i32 cpol).
// ---------------------------------------------------------------------------
static __device__ __forceinline__ void tdm_load_tile_f16(
    const _Float16* gsrc, unsigned lds_off, unsigned d0, unsigned rows) {
  unsigned long long ga = (unsigned long long)(uintptr_t)gsrc;
  v4u g0;
  g0[0] = 1u;                                        // count=1, user desc
  g0[1] = lds_off;                                   // lds_addr (bytes)
  g0[2] = (unsigned)(ga & 0xFFFFFFFFu);              // global_addr[31:0]
  g0[3] = (unsigned)((ga >> 32) & 0x1FFFFFFu)        // global_addr[56:32]
          | (2u << 30);                              // type = 2 (image)
  v8i g1;
  g1[0] = (int)((1u << 16)                           // data_size = 1 (2 bytes)
                | (1u << 20)                         // pad_enable
                | (7u << 22)                         // pad_interval: 256 DWORDs
                | (3u << 25));                       // pad_amount: 4 DWORDs
  g1[1] = (int)((d0 & 0xFFFFu) << 16);               // tensor_dim0[15:0]
  g1[2] = (int)((d0 >> 16) | ((rows & 0xFFFFu) << 16));   // d0 hi | tensor_dim1 lo
  g1[3] = (int)((rows >> 16) | ((d0 & 0xFFFFu) << 16));   // dim1 hi | tile_dim0
  g1[4] = (int)(rows & 0xFFFFu);                     // tile_dim1 = rows, tile_dim2=0
  g1[5] = (int)d0;                                   // tensor_dim0_stride[31:0]
  g1[6] = 0;                                         // stride0 hi, stride1 lo
  g1[7] = 0;
  v4i z4 = {0, 0, 0, 0};
  v8i z8 = {0, 0, 0, 0, 0, 0, 0, 0};
  __builtin_amdgcn_tensor_load_to_lds(g0, g1, z4, z4, z8, 0);
}

// ---------------------------------------------------------------------------
// Kernel 0: fp32 -> f16 conversion (one-shot; removes cvts from GEMM loops).
// ---------------------------------------------------------------------------
__global__ __launch_bounds__(256) void cvt_kernel(const float* __restrict__ in,
                                                  _Float16* __restrict__ out,
                                                  int n) {
  const int i = (blockIdx.x * 256 + threadIdx.x) * 8;
  if (i + 8 <= n) {
#pragma unroll
    for (int j = 0; j < 8; ++j) out[i + j] = (_Float16)in[i + j];
  }
}

// ---------------------------------------------------------------------------
// Kernel 1: QKV projection, all-f16 operands.  A-tile (16 x 512 f16) staged
// in LDS by TDM; B = Wh f16 row-major.  v_transposed: 0 -> out[bh][n][d],
// 1 -> out[bh][d][n].  grid.x = 512, block 256 (8 waves, 16x64 tile each).
// ---------------------------------------------------------------------------
__global__ __launch_bounds__(256) void proj_kernel(
    const _Float16* __restrict__ Xh, const _Float16* __restrict__ Wh,
    const float* __restrict__ bias, _Float16* __restrict__ out,
    int v_transposed) {
  __shared__ _Float16 Atile[16 * APAD];
  const int wave = threadIdx.x >> 5;
  const int lane = threadIdx.x & 31;
  const int rl = lane & 15, hi = lane >> 4;
  const int mtile = blockIdx.x;

  if (wave == 0) {
    tdm_load_tile_f16(Xh + (size_t)mtile * 16 * C_,
                      (unsigned)(uintptr_t)(void*)Atile, C_, 16);
    __builtin_amdgcn_s_wait_tensorcnt(0);
  }
  __syncthreads();

  v8f acc[4];
#pragma unroll
  for (int t = 0; t < 4; ++t) acc[t] = zero_v8f();

  for (int kt = 0; kt < C_ / 32; ++kt) {
    v16h a = load_a_f16(&Atile[rl * APAD + kt * 32 + hi * 16]);
#pragma unroll
    for (int t = 0; t < 4; ++t) {
      const int n = (wave * 4 + t) * 16 + rl;
      v16h b = load_b_f16(Wh + (size_t)n * C_ + kt * 32 + hi * 16);
      acc[t] = wmma_f16(a, b, acc[t]);
    }
  }

#pragma unroll
  for (int t = 0; t < 4; ++t) {
    const int col = (wave * 4 + t) * 16 + rl;
    const int h = col >> 6, d = col & 63;
    const float bb = bias[col];
#pragma unroll
    for (int r = 0; r < 8; ++r) {
      const int mm = mtile * 16 + hi * 8 + r;
      const int bidx = mm >> 10, nn = mm & (N_ - 1);
      const float v = acc[t][r] + bb;
      const size_t idx =
          v_transposed
              ? ((size_t)((bidx * H_ + h) * D_ + d) * N_ + nn)
              : ((size_t)((bidx * H_ + h) * N_ + nn) * D_ + d);
      out[idx] = (_Float16)v;
    }
  }
}

// ---------------------------------------------------------------------------
// Kernel 2 (fused): scores = QK^T/8 + adjacency bias + key mask; softmax;
// write attn_probs fp32; then context = P @ V computed from LDS-resident P
// (K-dim split over the 8 waves, partials summed via LDS), write ctx f16.
// grid = (N/16, BH), block 256.
// ---------------------------------------------------------------------------
__global__ __launch_bounds__(256) void attn_fused_kernel(
    const _Float16* __restrict__ qh, const _Float16* __restrict__ kh,
    const _Float16* __restrict__ vt, const float* __restrict__ adj,
    const int* __restrict__ mask, float* __restrict__ attn,
    _Float16* __restrict__ ctx) {
  __shared__ float S[16][SPAD];        // scores -> exp values
  __shared__ float Cpart[8][16][68];   // per-wave partial context
  __shared__ float red[16][16];
  __shared__ float stat[16];

  const int wave = threadIdx.x >> 5;
  const int lane = threadIdx.x & 31;
  const int rl = lane & 15, hi = lane >> 4;
  const int rt = blockIdx.x;  // row tile 0..63
  const int bh = blockIdx.y;  // 0..63
  const int b = bh >> 3, hh = bh & 7;

  // ---- Phase 1: scores via WMMA (Kdim=64 -> 2 k-steps) ----
  const _Float16* qrow = qh + ((size_t)bh * N_ + rt * 16 + rl) * D_;
  v16h a0 = load_a_f16(qrow + hi * 16);
  v16h a1 = load_a_f16(qrow + 32 + hi * 16);

  for (int i = 0; i < 8; ++i) {
    const int ct = wave * 8 + i;
    const _Float16* krow = kh + ((size_t)bh * N_ + ct * 16 + rl) * D_;
    v16h b0 = load_b_f16(krow + hi * 16);
    v16h b1 = load_b_f16(krow + 32 + hi * 16);
    v8f acc = zero_v8f();
    acc = wmma_f16(a0, b0, acc);
    acc = wmma_f16(a1, b1, acc);
#pragma unroll
    for (int r = 0; r < 8; ++r) {
      const int mrel = hi * 8 + r;
      const int key = ct * 16 + rl;
      float s = acc[r] * 0.125f;  // 1/sqrt(64)
      s += (1.0f - adj[(size_t)(rt * 16 + mrel) * N_ + key]) * NEGV;
      if (mask[b * N_ + key] == 0) s = NEGV;
      S[mrel][key] = s;
    }
  }
  __syncthreads();

  // ---- Phase 2: row max ----
  const int row = threadIdx.x >> 4;  // 0..15
  const int seg = threadIdx.x & 15;  // 0..15
  float mx = -3.0e38f;
  for (int c = 0; c < 64; ++c) mx = fmaxf(mx, S[row][c * 16 + seg]);
  red[row][seg] = mx;
  __syncthreads();
  if (seg == 0) {
    float m2 = red[row][0];
    for (int j = 1; j < 16; ++j) m2 = fmaxf(m2, red[row][j]);
    stat[row] = m2;
  }
  __syncthreads();

  // ---- Phase 3: exp + row sum (S keeps unnormalized exp) ----
  const float rowmax = stat[row];
  float sum = 0.f;
  for (int c = 0; c < 64; ++c) {
    const float e = __expf(S[row][c * 16 + seg] - rowmax);
    S[row][c * 16 + seg] = e;
    sum += e;
  }
  red[row][seg] = sum;
  __syncthreads();
  if (seg == 0) {
    float s2 = 0.f;
    for (int j = 0; j < 16; ++j) s2 += red[row][j];
    stat[row] = 1.0f / s2;
  }
  __syncthreads();

  // ---- Phase 4a: stream attn_probs to global (normalized) ----
  const float inv = stat[row];
  float* orow = attn + ((size_t)bh * N_ + rt * 16 + row) * N_;
  for (int c = 0; c < 64; ++c)
    orow[c * 16 + seg] = S[row][c * 16 + seg] * inv;

  // ---- Phase 4b: P @ V, K-dim split across waves (wave w: keys w*128..+128) ----
  v8f cacc[4];
#pragma unroll
  for (int t = 0; t < 4; ++t) cacc[t] = zero_v8f();
#pragma unroll
  for (int i = 0; i < 4; ++i) {
    const int ks = wave * 4 + i;  // k-step of 32 keys
    v16h a = load_a_f32(&S[rl][ks * 32 + hi * 16]);
#pragma unroll
    for (int t = 0; t < 4; ++t) {
      const int d = t * 16 + rl;
      v16h bfr = load_b_f16(vt + ((size_t)bh * D_ + d) * N_ + ks * 32 + hi * 16);
      cacc[t] = wmma_f16(a, bfr, cacc[t]);
    }
  }
#pragma unroll
  for (int t = 0; t < 4; ++t)
#pragma unroll
    for (int r = 0; r < 8; ++r)
      Cpart[wave][hi * 8 + r][t * 16 + rl] = cacc[t][r];
  __syncthreads();

  // ---- Phase 5: reduce partials, scale by 1/rowsum, store ctx f16 ----
#pragma unroll
  for (int i = 0; i < 4; ++i) {
    const int e = threadIdx.x + 256 * i;       // 0..1023
    const int er = e >> 6, ec = e & 63;
    float s2 = 0.f;
#pragma unroll
    for (int w = 0; w < 8; ++w) s2 += Cpart[w][er][ec];
    s2 *= stat[er];
    ctx[(size_t)(b * N_ + rt * 16 + er) * C_ + hh * D_ + ec] = (_Float16)s2;
  }
}

// ---------------------------------------------------------------------------
// Kernel 3: output = context @ Wo^T + bo (fp32 out).  A-tile via TDM to LDS,
// B = Woh f16.  grid.x = 512, block 256.
// ---------------------------------------------------------------------------
__global__ __launch_bounds__(256) void oproj_kernel(
    const _Float16* __restrict__ ctx, const _Float16* __restrict__ Woh,
    const float* __restrict__ bo, float* __restrict__ out) {
  __shared__ _Float16 Atile[16 * APAD];
  const int wave = threadIdx.x >> 5;
  const int lane = threadIdx.x & 31;
  const int rl = lane & 15, hi = lane >> 4;
  const int mtile = blockIdx.x;

  if (wave == 0) {
    tdm_load_tile_f16(ctx + (size_t)mtile * 16 * C_,
                      (unsigned)(uintptr_t)(void*)Atile, C_, 16);
    __builtin_amdgcn_s_wait_tensorcnt(0);
  }
  __syncthreads();

  v8f acc[4];
#pragma unroll
  for (int t = 0; t < 4; ++t) acc[t] = zero_v8f();

  for (int kt = 0; kt < C_ / 32; ++kt) {
    v16h a = load_a_f16(&Atile[rl * APAD + kt * 32 + hi * 16]);
#pragma unroll
    for (int t = 0; t < 4; ++t) {
      const int n = (wave * 4 + t) * 16 + rl;
      v16h b = load_b_f16(Woh + (size_t)n * C_ + kt * 32 + hi * 16);
      acc[t] = wmma_f16(a, b, acc[t]);
    }
  }

#pragma unroll
  for (int t = 0; t < 4; ++t) {
    const int col = (wave * 4 + t) * 16 + rl;
    const float bb = bo[col];
#pragma unroll
    for (int r = 0; r < 8; ++r) {
      const int mm = mtile * 16 + hi * 8 + r;
      out[(size_t)mm * C_ + col] = acc[t][r] + bb;
    }
  }
}

// ---------------------------------------------------------------------------
extern "C" void kernel_launch(void* const* d_in, const int* in_sizes, int n_in,
                              void* d_out, int out_size, void* d_ws,
                              size_t ws_size, hipStream_t stream) {
  (void)in_sizes; (void)n_in; (void)out_size; (void)ws_size;
  const float* x    = (const float*)d_in[0];
  const float* adj  = (const float*)d_in[1];
  const int*   mask = (const int*)d_in[2];
  const float* Wq   = (const float*)d_in[3];
  const float* bq   = (const float*)d_in[4];
  const float* Wk   = (const float*)d_in[5];
  const float* bk   = (const float*)d_in[6];
  const float* Wv   = (const float*)d_in[7];
  const float* bv   = (const float*)d_in[8];
  const float* Wo   = (const float*)d_in[9];
  const float* bo   = (const float*)d_in[10];

  float* out  = (float*)d_out;               // [B, N, C]
  float* attn = out + (size_t)B_ * N_ * C_;  // [B, H, N, N]

  const size_t headElems = (size_t)BH_ * N_ * D_;  // 4M
  const size_t wElems = (size_t)C_ * C_;           // 256K
  _Float16* qh  = (_Float16*)d_ws;                 //  8 MB
  _Float16* kh  = qh + headElems;                  //  8 MB
  _Float16* vt  = kh + headElems;                  //  8 MB  (V^T [bh][d][n])
  _Float16* ctx = vt + headElems;                  //  8 MB  ([B,N,C] f16)
  _Float16* xh  = ctx + headElems;                 //  8 MB  (x f16)
  _Float16* wqh = xh + (size_t)B_ * N_ * C_;       // 0.5 MB each
  _Float16* wkh = wqh + wElems;
  _Float16* wvh = wkh + wElems;
  _Float16* woh = wvh + wElems;

  dim3 blk(256);
  const int nx = B_ * N_ * C_;
  cvt_kernel<<<dim3(nx / 2048), blk, 0, stream>>>(x, xh, nx);
  cvt_kernel<<<dim3((int)wElems / 2048), blk, 0, stream>>>(Wq, wqh, (int)wElems);
  cvt_kernel<<<dim3((int)wElems / 2048), blk, 0, stream>>>(Wk, wkh, (int)wElems);
  cvt_kernel<<<dim3((int)wElems / 2048), blk, 0, stream>>>(Wv, wvh, (int)wElems);
  cvt_kernel<<<dim3((int)wElems / 2048), blk, 0, stream>>>(Wo, woh, (int)wElems);

  proj_kernel<<<dim3((B_ * N_) / 16), blk, 0, stream>>>(xh, wqh, bq, qh, 0);
  proj_kernel<<<dim3((B_ * N_) / 16), blk, 0, stream>>>(xh, wkh, bk, kh, 0);
  proj_kernel<<<dim3((B_ * N_) / 16), blk, 0, stream>>>(xh, wvh, bv, vt, 1);

  attn_fused_kernel<<<dim3(N_ / 16, BH_), blk, 0, stream>>>(qh, kh, vt, adj,
                                                            mask, attn, ctx);

  oproj_kernel<<<dim3((B_ * N_) / 16), blk, 0, stream>>>(ctx, woh, bo, out);
}
